// BrocaNetwork_7533372637661
// MI455X (gfx1250) — matrix operand order repaired and checked by reading, not compiled, not run on hardware
//
#include <hip/hip_runtime.h>
#include <hip/hip_bf16.h>

// ---------------------------------------------------------------------------
// Transformer decoder forward for MI455X (gfx1250, wave32, WMMA).
//   * All GEMMs run through v_wmma_f32_16x16x32_bf16 (D = A x B + C, f32 acc).
//   * Wave tile 32x64 = 2x4 WMMA tiles -> 8 independent WMMAs per K-step,
//     doubling A-fragment reuse and WMMA ILP vs a 32x32 tile.
//   * w_out converted once to bf16 (184 MB < 192 MB L2); GEMM grid is
//     M-fastest so each N-stripe of w_out is L2-resident across all M-tiles.
//   * Cross-attention has a single key (Sk==1) => softmax == 1 => it reduces
//     to two per-batch matvecs broadcast over sequence positions.
// ---------------------------------------------------------------------------

#define Bc 2
#define Sc 512
#define Dc 512
#define Hc 8
#define HDc 64
#define Lc 6
#define Vc 180000
#define Fc 2048
#define Mc (Bc * Sc)   // 1024 rows of activations

typedef __bf16 v16bf __attribute__((ext_vector_type(16)));
typedef __bf16 v8bf  __attribute__((ext_vector_type(8)));
typedef __bf16 v4bf  __attribute__((ext_vector_type(4)));
typedef float  v8f   __attribute__((ext_vector_type(8)));

// ---------------------------------------------------------------- cvt f32->bf16
__global__ void cvt_f32_bf16_kernel(const float* __restrict__ src,
                                    __hip_bfloat16* __restrict__ dst,
                                    long long n) {
  __bf16* d = reinterpret_cast<__bf16*>(dst);
  long long i4 = ((long long)blockIdx.x * blockDim.x + threadIdx.x) * 4;
  if (i4 + 3 < n) {
    float4 v = *(const float4*)(src + i4);
    v4bf o;
    o[0] = (__bf16)v.x; o[1] = (__bf16)v.y; o[2] = (__bf16)v.z; o[3] = (__bf16)v.w;
    *(v4bf*)(d + i4) = o;
  } else {
    for (long long j = i4; j < n; ++j) d[j] = (__bf16)src[j];
  }
}

// ------------------------------------------------------------------- embedding
__global__ void embed_kernel(const int* __restrict__ ids,
                             const float* __restrict__ emb,
                             const float* __restrict__ pos,
                             float* __restrict__ x) {
  int i = blockIdx.x * blockDim.x + threadIdx.x;       // M*D threads
  if (i >= Mc * Dc) return;
  int m = i / Dc, d = i % Dc;
  int s = m % Sc;
  x[i] = emb[(size_t)ids[m] * Dc + d] + pos[(size_t)s * Dc + d];
}

// ------------------------------------------------------- bf16 WMMA GEMM + bias
// C(M,N) = A(M,K) @ W(N,K)^T + bias, optional ReLU.
// Wave tile 32(M) x 64(N) = 2x4 WMMA 16x16x32 tiles.
// Block = 8 waves arranged 4(M) x 2(N) -> 128 x 128 block tile.
__global__ __launch_bounds__(256) void gemm_bf16_kernel(
    const __hip_bfloat16* __restrict__ Ap,
    const __hip_bfloat16* __restrict__ Wp,
    const float* __restrict__ bias,
    float* __restrict__ C,
    int Mdim, int Ndim, int Kdim, int relu) {
  const __bf16* A = reinterpret_cast<const __bf16*>(Ap);
  const __bf16* W = reinterpret_cast<const __bf16*>(Wp);

  int lane  = threadIdx.x & 31;
  int wave  = threadIdx.x >> 5;
  int waveM = wave & 3;          // 4 waves along M
  int waveN = wave >> 2;         // 2 waves along N
  int mbase = blockIdx.x * 128 + waveM * 32;
  int nbase = blockIdx.y * 128 + waveN * 64;

  v8f acc[2][4] = {};

  // A fragment (16x32 bf16): lanes 0-15 rows M0..15 w/ K 0..15,
  // lanes 16-31 same rows w/ K 16..31  -> two 16B loads.
  int arow0 = mbase + (lane & 15);
  int ahalf = (lane >> 4) << 4;           // 0 or 16
  // B fragment (32x16 bf16): lanes 0-15 col N0..15 K0..15, lanes 16-31 K16..31
  int bcol0 = nbase + (lane & 15);
  int bkoff = (lane >> 4) << 4;           // 0 or 16

  for (int k0 = 0; k0 < Kdim; k0 += 32) {
    if (k0 + 32 < Kdim) {                 // gfx1250 global_prefetch_b8
      __builtin_prefetch(A + (size_t)arow0 * Kdim + k0 + 32, 0, 1);
      __builtin_prefetch(W + (size_t)bcol0 * Kdim + k0 + 32, 0, 1);
    }
    v16bf a[2], b[4];
#pragma unroll
    for (int i = 0; i < 2; ++i) {
      int r = arow0 + i * 16; if (r >= Mdim) r = Mdim - 1;
      const __bf16* p = A + (size_t)r * Kdim + k0 + ahalf;
      v8bf lo = *(const v8bf*)p;
      v8bf hi = *(const v8bf*)(p + 8);
#pragma unroll
      for (int e = 0; e < 8; ++e) { a[i][e] = lo[e]; a[i][e + 8] = hi[e]; }
    }
#pragma unroll
    for (int j = 0; j < 4; ++j) {
      int c = bcol0 + j * 16; if (c >= Ndim) c = Ndim - 1;
      b[j] = *(const v16bf*)(W + (size_t)c * Kdim + k0 + bkoff);
    }
#pragma unroll
    for (int i = 0; i < 2; ++i)
#pragma unroll
      for (int j = 0; j < 4; ++j)
        acc[i][j] = __builtin_amdgcn_wmma_f32_16x16x32_bf16(
            false, a[i], false, b[j], (short)0, acc[i][j], false, false);
  }

  // C/D layout: VGPR r -> M=r (lanes 0-15) / M=r+8 (lanes 16-31), N=lane&15.
#pragma unroll
  for (int i = 0; i < 2; ++i) {
    int mrow = mbase + i * 16 + ((lane >> 4) << 3);
#pragma unroll
    for (int j = 0; j < 4; ++j) {
      int n = nbase + j * 16 + (lane & 15);
      if (n < Ndim) {
        float bi = bias ? bias[n] : 0.0f;
#pragma unroll
        for (int r = 0; r < 8; ++r) {
          int m = mrow + r;
          if (m < Mdim) {
            float v = acc[i][j][r] + bi;
            if (relu) v = fmaxf(v, 0.0f);
            C[(size_t)m * Ndim + n] = v;
          }
        }
      }
    }
  }
}

// -------------------------------------------------- causal attention (per q)
__global__ void attn_kernel(const float* __restrict__ Q,
                            const float* __restrict__ K,
                            const float* __restrict__ V,
                            float* __restrict__ O) {
  int t = blockIdx.x * blockDim.x + threadIdx.x;   // B*H*S threads, q fastest
  if (t >= Bc * Hc * Sc) return;
  int q = t % Sc;
  int h = (t / Sc) % Hc;
  int b = t / (Sc * Hc);

  const float* qp = Q + ((size_t)(b * Sc + q) * Dc) + h * HDc;
  float qv[HDc];
#pragma unroll
  for (int d = 0; d < HDc; ++d) qv[d] = qp[d] * 0.125f;   // 1/sqrt(64)

  float mx = -3.4e38f, sum = 0.0f;
  float acc[HDc];
#pragma unroll
  for (int d = 0; d < HDc; ++d) acc[d] = 0.0f;

  for (int kk = 0; kk <= q; ++kk) {                 // causal
    const float* kp = K + ((size_t)(b * Sc + kk) * Dc) + h * HDc;
    float dot = 0.0f;
#pragma unroll
    for (int d = 0; d < HDc; ++d) dot = fmaf(qv[d], kp[d], dot);
    float nm = fmaxf(mx, dot);
    float sc = __expf(mx - nm);
    float w  = __expf(dot - nm);
    sum = sum * sc + w;
    const float* vp = V + ((size_t)(b * Sc + kk) * Dc) + h * HDc;
#pragma unroll
    for (int d = 0; d < HDc; ++d) acc[d] = fmaf(w, vp[d], acc[d] * sc);
    mx = nm;
  }
  float inv = 1.0f / sum;
  float* op = O + ((size_t)(b * Sc + q) * Dc) + h * HDc;
#pragma unroll
  for (int d = 0; d < HDc; ++d) op[d] = acc[d] * inv;
}

// ------------------------------------------------ x = LN(x + h) (D=512 row)
// bcastS==0: h row == x row;  bcastS==Sc: h row = row / Sc (per-batch bcast)
__global__ void add_ln_kernel(float* __restrict__ x,
                              const float* __restrict__ h,
                              const float* __restrict__ g,
                              const float* __restrict__ bb,
                              int bcastS) {
  int row = blockIdx.x, t = threadIdx.x;
  const float* hr = h + (size_t)(bcastS ? (row / bcastS) : row) * Dc;
  float* xr = x + (size_t)row * Dc;
  float a0 = xr[t]       + hr[t];
  float a1 = xr[t + 256] + hr[t + 256];
  __shared__ float rs[256], rq[256];
  rs[t] = a0 + a1;
  rq[t] = a0 * a0 + a1 * a1;
  __syncthreads();
  for (int off = 128; off > 0; off >>= 1) {
    if (t < off) { rs[t] += rs[t + off]; rq[t] += rq[t + off]; }
    __syncthreads();
  }
  float mean = rs[0] * (1.0f / Dc);
  float var  = rq[0] * (1.0f / Dc) - mean * mean;
  float inv  = rsqrtf(var + 1e-5f);
  xr[t]       = (a0 - mean) * inv * g[t]       + bb[t];
  xr[t + 256] = (a1 - mean) * inv * g[t + 256] + bb[t + 256];
}

// ---------------------------------------- tiny matvec: y(B,N) = x(B,K)@w^T+b
__global__ void matvec_kernel(const float* __restrict__ xin,
                              const float* __restrict__ w,
                              const float* __restrict__ bias,
                              float* __restrict__ y,
                              int Bn, int N, int Kd) {
  int idx = blockIdx.x * blockDim.x + threadIdx.x;
  if (idx >= Bn * N) return;
  int b = idx / N, n = idx % N;
  const float* xr = xin + (size_t)b * Kd;
  const float* wr = w + (size_t)n * Kd;
  float dot = 0.0f;
  for (int k = 0; k < Kd; ++k) dot = fmaf(xr[k], wr[k], dot);
  y[idx] = dot + bias[n];
}

// ---------------------------------------------------------------------------
extern "C" void kernel_launch(void* const* d_in, const int* in_sizes, int n_in,
                              void* d_out, int out_size, void* d_ws, size_t ws_size,
                              hipStream_t stream) {
  (void)in_sizes; (void)n_in; (void)out_size; (void)ws_size;

  const float* meaning = (const float*)d_in[0];
  const int*   ids     = (const int*)  d_in[1];
  const float* emb     = (const float*)d_in[2];
  const float* pos     = (const float*)d_in[3];
  const float* sa_wq = (const float*)d_in[4],  *sa_wk = (const float*)d_in[5];
  const float* sa_wv = (const float*)d_in[6],  *sa_bq = (const float*)d_in[7];
  const float* sa_bk = (const float*)d_in[8],  *sa_bv = (const float*)d_in[9];
  const float* sa_wo = (const float*)d_in[10], *sa_bo = (const float*)d_in[11];
  // ca_wq/ca_wk/ca_bq/ca_bk (12,13,15,16) are dead: softmax over 1 key == 1.
  const float* ca_wv = (const float*)d_in[14];
  const float* ca_bv = (const float*)d_in[17];
  const float* ca_wo = (const float*)d_in[18], *ca_bo = (const float*)d_in[19];
  const float* ln1_g = (const float*)d_in[20], *ln1_b = (const float*)d_in[21];
  const float* ln2_g = (const float*)d_in[22], *ln2_b = (const float*)d_in[23];
  const float* ln3_g = (const float*)d_in[24], *ln3_b = (const float*)d_in[25];
  const float* ff_w1 = (const float*)d_in[26], *ff_b1 = (const float*)d_in[27];
  const float* ff_w2 = (const float*)d_in[28], *ff_b2 = (const float*)d_in[29];
  const float* w_out = (const float*)d_in[30], *b_out = (const float*)d_in[31];

  // -------- workspace carve-up (256B aligned) --------
  char* ws = (char*)d_ws;
  size_t off = 0;
  auto take = [&](size_t bytes) -> char* {
    char* p = ws + off;
    off = (off + bytes + 255) & ~(size_t)255;
    return p;
  };
  float* X   = (float*)take((size_t)Mc * Dc * 4);   // activations
  float* Hb  = (float*)take((size_t)Mc * Dc * 4);   // residual branch
  float* Qb  = (float*)take((size_t)Mc * Dc * 4);
  float* Kb  = (float*)take((size_t)Mc * Dc * 4);
  float* Vb  = (float*)take((size_t)Mc * Dc * 4);
  float* AO  = (float*)take((size_t)Mc * Dc * 4);   // attention output
  float* FFb = (float*)take((size_t)Mc * Fc * 4);
  float* CVb = (float*)take((size_t)Bc * Dc * 4);
  float* COb = (float*)take((size_t)Bc * Dc * 4);
  __hip_bfloat16* Abf = (__hip_bfloat16*)take((size_t)Mc * Fc * 2);
  __hip_bfloat16* Wbf = (__hip_bfloat16*)take((size_t)Vc * Dc * 2);  // 184 MB

  auto cvt = [&](const float* s, __hip_bfloat16* d, size_t n) {
    long long nv = ((long long)n + 3) / 4;
    int blocks = (int)((nv + 255) / 256);
    cvt_f32_bf16_kernel<<<blocks, 256, 0, stream>>>(s, d, (long long)n);
  };
  auto gemm = [&](const __hip_bfloat16* Aa, const __hip_bfloat16* Ww,
                  const float* bias, float* Cc, int Mm, int Nn, int Kk, int relu) {
    dim3 grid((Mm + 127) / 128, (Nn + 127) / 128);  // M fastest -> L2 reuse of W
    gemm_bf16_kernel<<<grid, 256, 0, stream>>>(Aa, Ww, bias, Cc, Mm, Nn, Kk, relu);
  };

  // -------- embedding + positions --------
  embed_kernel<<<(Mc * Dc + 255) / 256, 256, 0, stream>>>(ids, emb, pos, X);

  // -------- layers --------
  for (int l = 0; l < Lc; ++l) {
    size_t wofs = (size_t)l * Dc * Dc;
    size_t bofs = (size_t)l * Dc;

    // self-attention
    cvt(X, Abf, (size_t)Mc * Dc);
    cvt(sa_wq + wofs, Wbf, (size_t)Dc * Dc);
    gemm(Abf, Wbf, sa_bq + bofs, Qb, Mc, Dc, Dc, 0);
    cvt(sa_wk + wofs, Wbf, (size_t)Dc * Dc);
    gemm(Abf, Wbf, sa_bk + bofs, Kb, Mc, Dc, Dc, 0);
    cvt(sa_wv + wofs, Wbf, (size_t)Dc * Dc);
    gemm(Abf, Wbf, sa_bv + bofs, Vb, Mc, Dc, Dc, 0);
    attn_kernel<<<(Bc * Hc * Sc + 255) / 256, 256, 0, stream>>>(Qb, Kb, Vb, AO);
    cvt(AO, Abf, (size_t)Mc * Dc);
    cvt(sa_wo + wofs, Wbf, (size_t)Dc * Dc);
    gemm(Abf, Wbf, sa_bo + bofs, Hb, Mc, Dc, Dc, 0);
    add_ln_kernel<<<Mc, 256, 0, stream>>>(X, Hb, ln1_g + bofs, ln1_b + bofs, 0);

    // cross-attention: Sk==1 -> softmax==1 -> (mem@Wv^T+bv)@Wo^T+bo, bcast
    matvec_kernel<<<(Bc * Dc + 255) / 256, 256, 0, stream>>>(
        meaning, ca_wv + wofs, ca_bv + bofs, CVb, Bc, Dc, Dc);
    matvec_kernel<<<(Bc * Dc + 255) / 256, 256, 0, stream>>>(
        CVb, ca_wo + wofs, ca_bo + bofs, COb, Bc, Dc, Dc);
    add_ln_kernel<<<Mc, 256, 0, stream>>>(X, COb, ln2_g + bofs, ln2_b + bofs, Sc);

    // FFN
    cvt(X, Abf, (size_t)Mc * Dc);
    cvt(ff_w1 + (size_t)l * Fc * Dc, Wbf, (size_t)Fc * Dc);
    gemm(Abf, Wbf, ff_b1 + (size_t)l * Fc, FFb, Mc, Fc, Dc, 1);   // +ReLU
    cvt(FFb, Abf, (size_t)Mc * Fc);
    cvt(ff_w2 + (size_t)l * Dc * Fc, Wbf, (size_t)Dc * Fc);
    gemm(Abf, Wbf, ff_b2 + bofs, Hb, Mc, Dc, Fc, 0);
    add_ln_kernel<<<Mc, 256, 0, stream>>>(X, Hb, ln3_g + bofs, ln3_b + bofs, 0);
  }

  // -------- final vocab projection (dominant GEMM; Wbf stripe lives in L2) --
  cvt(X, Abf, (size_t)Mc * Dc);
  cvt(w_out, Wbf, (size_t)Vc * Dc);
  gemm(Abf, Wbf, b_out, (float*)d_out, Mc, Vc, Dc, 0);
}
